// RNNRegressor_37787122270217
// MI455X (gfx1250) — matrix operand order repaired and checked by reading, not compiled, not run on hardware
//
#include <hip/hip_runtime.h>
#include <hip/hip_bf16.h>
#include <math.h>

typedef __bf16 bf16_t;
typedef __attribute__((ext_vector_type(16))) __bf16 v16bf;
typedef __attribute__((ext_vector_type(8)))  float  v8f;

// Problem constants: B=64, T=512, U=256, 4U=1024, NOUT=4
#define TT   512
#define UU   256
#define GG   1024   // 4*U
#define BB   64

// Fragment union for building WMMA A operands from scalar loads.
union AFrag {
    v16bf v;
    bf16_t e[16];
    unsigned int w[8];
};

// ---------------------------------------------------------------------------
// Weight packing: fp32 [K=256, N=1024] row-major -> bf16 B-fragment layout.
// Fragment (nt, kc) = 32x16 tile. Per ISA B layout (16-bit, 32x16): lane l
// holds column n = l&15; the 16 halfs j of lane half h = l>>4
// are K offsets (j>>3)*16 + h*8 + (j&7) within the 32-K chunk.
// packed[((nt*8+kc)*32 + lane)*16 + j] = B[kc*32 + (j>>3)*16 + (lane>>4)*8 + (j&7)][nt*16 + (lane&15)]
// ---------------------------------------------------------------------------
__global__ void pack_w(const float* __restrict__ src, bf16_t* __restrict__ dst) {
    int t = blockIdx.x * blockDim.x + threadIdx.x;   // 0 .. 262143 (exact)
    int j    = t & 15;
    int lane = (t >> 4) & 31;
    int kc   = (t >> 9) & 7;
    int nt   = t >> 12;                              // 0..63
    int k = kc * 32 + ((j >> 3) << 4) + ((lane >> 4) << 3) + (j & 7);
    int n = nt * 16 + (lane & 15);
    dst[t] = (bf16_t)src[(size_t)k * GG + n];
}

// ---------------------------------------------------------------------------
// Input projection GEMM: out = A[M=32768, 256] (fp32) x Wpacked(bf16) + bias.
// Output written in scan-friendly swizzled layout:
//   xz[((b>>4)*512 + t)*16384 + (b&15)*1024 + n],  flat row r = b*512 + t.
// One wave computes a 16(M) x 64(N) tile: 4 n-tiles, 8 k-chunks, 32 WMMAs.
// blockDim=512 (16 waves), gridDim=2048 -> 32768 wave-tasks.
// ---------------------------------------------------------------------------
__global__ void gemm_xz(const float* __restrict__ A, const bf16_t* __restrict__ Wp,
                        const float* __restrict__ bias, float* __restrict__ xz) {
    const int wv   = threadIdx.x >> 5;
    const int lane = threadIdx.x & 31;
    const int lm   = lane & 15;
    const int lhi  = lane >> 4;

    const int task = blockIdx.x * 16 + wv;   // 0..32767
    const int mt   = task >> 4;              // m-tile 0..2047
    const int ng   = task & 15;              // n-group 0..15 (4 tiles each)

    const v16bf* __restrict__ Bp = (const v16bf*)Wp;

    v8f acc[4];
#pragma unroll
    for (int q = 0; q < 4; ++q) {
        float bv = bias[(ng * 4 + q) * 16 + lm];
#pragma unroll
        for (int v = 0; v < 8; ++v) acc[q][v] = bv;
    }

    const int row = mt * 16 + lm;            // A row this lane supplies
#pragma unroll
    for (int kc = 0; kc < 8; ++kc) {
        AFrag a;
#pragma unroll
        for (int v = 0; v < 8; ++v) {
            // A layout (16-bit 16x32): VGPR v holds K = (v>=4)*16 + (lane>>4)*8 + (v&3)*2 {+0,+1}
            int k = kc * 32 + ((v & 4) << 2) + lhi * 8 + (v & 3) * 2;
            const float2 f = *(const float2*)(A + (size_t)row * UU + k);
            a.e[2 * v]     = (bf16_t)f.x;
            a.e[2 * v + 1] = (bf16_t)f.y;
        }
        v16bf b[4];
#pragma unroll
        for (int q = 0; q < 4; ++q)
            b[q] = Bp[((ng * 4 + q) * 8 + kc) * 32 + lane];
#pragma unroll
        for (int q = 0; q < 4; ++q)
            acc[q] = __builtin_amdgcn_wmma_f32_16x16x32_bf16(
                false, a.v, false, b[q], (short)0, acc[q], false, false);
    }

    // Store: C/D layout -> lane col n = lane&15, VGPR v -> m = (lane>>4)*8 + v.
#pragma unroll
    for (int q = 0; q < 4; ++q) {
        int ncol = (ng * 4 + q) * 16 + lm;
#pragma unroll
        for (int v = 0; v < 8; ++v) {
            int rr = mt * 16 + lhi * 8 + v;          // flat row = b*512 + t
            int b_ = rr >> 9;
            int t_ = rr & 511;
            size_t oi = ((size_t)(b_ >> 4) * TT + t_) * (16 * GG)
                      + (size_t)(b_ & 15) * GG + ncol;
            xz[oi] = acc[q][v];
        }
    }
}

// ---------------------------------------------------------------------------
// Persistent LSTM scan. 4 workgroups x 512 threads (16 wave32s); workgroup w
// owns batch rows 16w..16w+15 and runs all 512 steps.
//
// Gate-aligned wave mapping: wave wv owns unit columns u0 = wv*16..wv*16+15
// in ALL FOUR gate blocks (N-tiles wv, wv+16, wv+32, wv+48). After the WMMAs,
// acc[0..3] hold z_i, z_f, z_g, z_o for matching (m,u) per lane/VGPR, so the
// gate nonlinearities run entirely in registers and the cell state c lives in
// a register v8f across all steps. Only h round-trips through LDS (bf16,
// double-buffered) for the next step's A fragments -> ONE barrier per step.
// ---------------------------------------------------------------------------
__global__ void lstm_scan(const float* __restrict__ xz, const bf16_t* __restrict__ Wp,
                          const float* __restrict__ resid, float* __restrict__ hseq,
                          float* __restrict__ hlast) {
    __shared__ bf16_t hBuf[2][16 * UU];              // 2 x 8KB double buffer

    const int tid  = threadIdx.x;
    const int wv   = tid >> 5;
    const int lane = tid & 31;
    const int lm   = lane & 15;
    const int lhi  = lane >> 4;
    const int m0   = blockIdx.x * 16;
    const int u    = wv * 16 + lm;                   // unit column this lane owns

    const v16bf* __restrict__ Bp = (const v16bf*)Wp;

    v8f cReg;                                        // cell state, register-resident
#pragma unroll
    for (int v = 0; v < 8; ++v) cReg[v] = 0.0f;

    for (int i = tid; i < 16 * UU; i += 512) hBuf[0][i] = (bf16_t)0.0f;
    __syncthreads();

    for (int t = 0; t < TT; ++t) {
        const bf16_t* hb = hBuf[t & 1];
        bf16_t*       hn = hBuf[(t + 1) & 1];
        const float* __restrict__ xzb =
            xz + ((size_t)blockIdx.x * TT + t) * (16 * GG);

        // ---- init accumulators from xz: acc[g] covers columns g*256 + u ----
        v8f acc[4];
#pragma unroll
        for (int q = 0; q < 4; ++q) {
            int ncol = q * 256 + u;
#pragma unroll
            for (int v = 0; v < 8; ++v)
                acc[q][v] = xzb[(size_t)(lhi * 8 + v) * GG + ncol];
        }

        // ---- z += h @ Uh via WMMA (A from LDS bf16 h, B from L2 weights) ----
#pragma unroll
        for (int kc = 0; kc < 8; ++kc) {
            AFrag a;
#pragma unroll
            for (int v = 0; v < 8; ++v) {
                int k = kc * 32 + ((v & 4) << 2) + lhi * 8 + (v & 3) * 2;
                a.w[v] = *(const unsigned int*)&hb[lm * UU + k];  // ds_load_b32
            }
            v16bf b[4];
#pragma unroll
            for (int q = 0; q < 4; ++q)
                b[q] = Bp[((q * 16 + wv) * 8 + kc) * 32 + lane];
#pragma unroll
            for (int q = 0; q < 4; ++q)
                acc[q] = __builtin_amdgcn_wmma_f32_16x16x32_bf16(
                    false, a.v, false, b[q], (short)0, acc[q], false, false);
        }

        // ---- gates: i,f,g,o all in registers; update c (regs) and h ----
#pragma unroll
        for (int v = 0; v < 8; ++v) {
            int m = lhi * 8 + v;
            float ig = 1.0f / (1.0f + expf(-acc[0][v]));
            float fg = 1.0f / (1.0f + expf(-acc[1][v]));
            float gg = tanhf(acc[2][v]);
            float og = 1.0f / (1.0f + expf(-acc[3][v]));
            float c  = fg * cReg[v] + ig * gg;
            float h  = og * tanhf(c);
            cReg[v] = c;
            hn[m * UU + u] = (bf16_t)h;              // h for next step (other buffer)
            if (hseq) {
                size_t oi = ((size_t)(m0 + m) * TT + t) * UU + u;
                float r = resid ? resid[oi] : 0.0f;
                hseq[oi] = h + r;
            }
            if (hlast && t == TT - 1)
                hlast[(size_t)(m0 + m) * UU + u] = h;
        }
        __syncthreads();                             // single barrier per step
    }
}

// ---------------------------------------------------------------------------
// Final dense head: out[m,k] = sum_{t,u} (h1[m,t,u] + h2[m,u]) * Wd[t*U+u, k] + bd[k]
// One block per batch row m; 256 threads; thread tid owns unit u = tid.
// ---------------------------------------------------------------------------
__global__ void final_dense(const float* __restrict__ h1, const float* __restrict__ h2,
                            const float* __restrict__ Wd, const float* __restrict__ bd,
                            float* __restrict__ out) {
    const int m   = blockIdx.x;
    const int tid = threadIdx.x;
    float h2v = h2[(size_t)m * UU + tid];
    float a0 = 0.f, a1 = 0.f, a2 = 0.f, a3 = 0.f;
    for (int t = 0; t < TT; ++t) {
        float hv = h1[((size_t)m * TT + t) * UU + tid] + h2v;
        const float4 w = *(const float4*)(Wd + ((size_t)t * UU + tid) * 4);
        a0 += hv * w.x; a1 += hv * w.y; a2 += hv * w.z; a3 += hv * w.w;
    }
    __shared__ float red[4][256];
    red[0][tid] = a0; red[1][tid] = a1; red[2][tid] = a2; red[3][tid] = a3;
    __syncthreads();
    for (int s = 128; s > 0; s >>= 1) {
        if (tid < s) {
#pragma unroll
            for (int k = 0; k < 4; ++k) red[k][tid] += red[k][tid + s];
        }
        __syncthreads();
    }
    if (tid < 4) out[m * 4 + tid] = red[tid][0] + bd[tid];
}

// ---------------------------------------------------------------------------
extern "C" void kernel_launch(void* const* d_in, const int* in_sizes, int n_in,
                              void* d_out, int out_size, void* d_ws, size_t ws_size,
                              hipStream_t stream) {
    (void)in_sizes; (void)n_in; (void)out_size; (void)ws_size;

    const float* x  = (const float*)d_in[0];
    const float* Wr = (const float*)d_in[1];
    const float* Ur = (const float*)d_in[2];
    const float* br = (const float*)d_in[3];
    const float* Wn = (const float*)d_in[4];
    const float* Un = (const float*)d_in[5];
    const float* bn = (const float*)d_in[6];
    const float* Wd = (const float*)d_in[7];
    const float* bd = (const float*)d_in[8];
    float* out = (float*)d_out;

    char* ws = (char*)d_ws;
    const size_t PK = (size_t)UU * GG * sizeof(bf16_t);   // 512 KB per packed matrix
    bf16_t* WrP = (bf16_t*)(ws + 0 * PK);
    bf16_t* UrP = (bf16_t*)(ws + 1 * PK);
    bf16_t* WnP = (bf16_t*)(ws + 2 * PK);
    bf16_t* UnP = (bf16_t*)(ws + 3 * PK);
    float* xzbuf = (float*)(ws + 4 * PK);                                  // 128 MB
    float* h0    = xzbuf + (size_t)BB * TT * GG;                           //  32 MB
    float* h1    = h0    + (size_t)BB * TT * UU;                           //  32 MB
    float* h2    = h1    + (size_t)BB * TT * UU;                           //  64 KB

    // 1) Pack weights into bf16 WMMA B-fragment layout.
    pack_w<<<1024, 256, 0, stream>>>(Wr, WrP);
    pack_w<<<1024, 256, 0, stream>>>(Ur, UrP);
    pack_w<<<1024, 256, 0, stream>>>(Wn, WnP);
    pack_w<<<1024, 256, 0, stream>>>(Un, UnP);

    // 2) Layer 0: xz = x @ Wr + br ; scan -> h0
    gemm_xz<<<2048, 512, 0, stream>>>(x, WrP, br, xzbuf);
    lstm_scan<<<4, 512, 0, stream>>>(xzbuf, UrP, nullptr, h0, nullptr);

    // 3) Layer 1 (same shared weights): xz = h0 @ Wr + br ; h1 = lstm(h0) + h0
    gemm_xz<<<2048, 512, 0, stream>>>(h0, WrP, br, xzbuf);
    lstm_scan<<<4, 512, 0, stream>>>(xzbuf, UrP, h0, h1, nullptr);

    // 4) Layer 2: xz = h1 @ Wn + bn ; scan -> last h only (h2)
    gemm_xz<<<2048, 512, 0, stream>>>(h1, WnP, bn, xzbuf);
    lstm_scan<<<4, 512, 0, stream>>>(xzbuf, UnP, nullptr, nullptr, h2);

    // 5) Head: y = flatten(h2[:,None,:] + h1) @ Wd + bd
    final_dense<<<64, 256, 0, stream>>>(h1, h2, Wd, bd, out);
}
